// Corr_13091060318904
// MI455X (gfx1250) — matrix operand order repaired and checked by reading, not compiled, not run on hardware
//
#include <hip/hip_runtime.h>
#include <hip/hip_bf16.h>

// ---------------------------------------------------------------------------
// MI455X (gfx1250) fused local-attention refinement.
//   k_gemm     : 1x1 convs as bf16 WMMA GEMM  (v_wmma_f32_16x16x32_bf16)
//                - B tiles (weights) register-resident, loaded once per wave
//                - bias pre-seeded into the WMMA C accumulator
//                - 4 pixel-tiles per wave -> 16 WMMAs per wave job
//   k_upsample : align-corners bilinear 64->128
//   k_attn     : 9x9 correlation + online softmax + weighted aggregation
// Intermediates (f1,f2,out_up ~8.6MB) live in d_ws and stay L2-resident
// (192 MB L2), so the 3-kernel split costs ~nothing vs. HBM roofline
// (~39 MB mandatory traffic @ 23.3 TB/s ≈ 1.7 us floor).
// ---------------------------------------------------------------------------

typedef __attribute__((ext_vector_type(16))) __bf16 v16bf;
typedef __attribute__((ext_vector_type(8)))  float  v8f;

static constexpr int N_   = 4;
static constexpr int Hh   = 128;
static constexpr int Ww   = 128;
static constexpr int HWp  = Hh * Ww;      // 16384
static constexpr int CIN  = 64;
static constexpr int CMID = 16;

// ---------------------------------------------------------------------------
// Kernel 1: f1 = delta x w1 + b1 ; f2 = delta x w2 + b2
// GEMM view per WMMA: A = [16 px x 32 ch] bf16, B = [32 ch x 16 out] bf16,
// D/C = [16 px x 16 out] f32.  One wave sweeps 4 pixel-tiles (64 px).
// ---------------------------------------------------------------------------
__global__ __launch_bounds__(256) void k_gemm(
    const float* __restrict__ delta,
    const float* __restrict__ w1, const float* __restrict__ b1,
    const float* __restrict__ w2, const float* __restrict__ b2,
    float* __restrict__ f1, float* __restrict__ f2) {
  const int lane = threadIdx.x & 31;
  const int wave = threadIdx.x >> 5;
  const int job  = blockIdx.x * 8 + wave;          // 1024 jobs total
  const int p0   = job * 64;                       // 64 pixels per wave
  const int n    = p0 >> 14;                       // HW = 16384
  const int ph0  = p0 & (HWp - 1);

  const int m    = lane & 15;                      // A row  (pixel in tile)
  const int half = lane >> 4;                      // lane half selects K block
  const int o    = lane & 15;                      // B col / D col (out chan)

  // ---- B tiles: loaded ONCE, register-resident across all pixel-tiles ----
  // 32x16 bf16, contiguous K (same as A): N = lane%16, lane-half carries K block of 16,
  // VGPR j holds K = half*16 + 2j, 2j+1.
  v16bf bA[2], bB[2];
#pragma unroll
  for (int s = 0; s < 2; ++s) {
#pragma unroll
    for (int j = 0; j < 8; ++j) {
      const int c = s * 32 + half * 16 + 2 * j;
      bA[s][2 * j]     = (__bf16)w1[o * CIN + c];
      bA[s][2 * j + 1] = (__bf16)w1[o * CIN + c + 1];
      bB[s][2 * j]     = (__bf16)w2[o * CIN + c];
      bB[s][2 * j + 1] = (__bf16)w2[o * CIN + c + 1];
    }
  }
  const float bias1 = b1[o];
  const float bias2 = b2[o];

  const float* dbase = delta + ((size_t)n * CIN) * HWp + (ph0 + m);
  float* f1p = f1 + ((size_t)n * CMID + o) * HWp + ph0;
  float* f2p = f2 + ((size_t)n * CMID + o) * HWp + ph0;

#pragma unroll
  for (int t = 0; t < 4; ++t) {
    // Seed C with bias: lane's D column N = o is fixed over all 8 VGPRs.
    v8f acc1, acc2;
#pragma unroll
    for (int r = 0; r < 8; ++r) { acc1[r] = bias1; acc2[r] = bias2; }

    const float* dp = dbase + t * 16;
#pragma unroll
    for (int s = 0; s < 2; ++s) {
      // ---- A tile: 16x32 bf16, contiguous K (same as B) ----------------------------
      // VGPR j   (elems 2j,2j+1)  : K = half*16 + 2j      (j = 0..3)
      // VGPR 4+j (elems 8+2j,...) : K = half*16 + 8 + 2j
      v16bf a;
      const int cb = s * 32 + half * 16;
#pragma unroll
      for (int j = 0; j < 4; ++j) {
        a[2 * j]         = (__bf16)dp[(size_t)(cb + 2 * j)          * HWp];
        a[2 * j + 1]     = (__bf16)dp[(size_t)(cb + 2 * j + 1)      * HWp];
        a[8 + 2 * j]     = (__bf16)dp[(size_t)(cb + 8 + 2 * j)     * HWp];
        a[8 + 2 * j + 1] = (__bf16)dp[(size_t)(cb + 8 + 2 * j + 1) * HWp];
      }
      acc1 = __builtin_amdgcn_wmma_f32_16x16x32_bf16(
          false, a, false, bA[s], (short)0, acc1, false, false);
      acc2 = __builtin_amdgcn_wmma_f32_16x16x32_bf16(
          false, a, false, bB[s], (short)0, acc2, false, false);
    }

    // D layout: VGPR r -> M = r + 8*half, N = lane%16
#pragma unroll
    for (int r = 0; r < 8; ++r) {
      const int mrow = t * 16 + r + 8 * half;
      f1p[mrow] = acc1[r];
      f2p[mrow] = acc2[r];
    }
  }
}

// ---------------------------------------------------------------------------
// Kernel 2: bilinear align-corners upsample out [N,2,64,64] -> [N,2,128,128]
// ---------------------------------------------------------------------------
__global__ __launch_bounds__(256) void k_upsample(
    const float* __restrict__ outs, float* __restrict__ oup) {
  const int idx = blockIdx.x * 256 + threadIdx.x;        // 131072 total
  const int x = idx & 127;
  const int y = (idx >> 7) & 127;
  const int c = (idx >> 14) & 1;
  const int n = idx >> 15;
  const float s  = 63.0f / 127.0f;
  const float fy = y * s, fx = x * s;
  const int y0 = (int)fy, x0 = (int)fx;
  const int y1 = min(y0 + 1, 63), x1 = min(x0 + 1, 63);
  const float wy = fy - (float)y0, wx = fx - (float)x0;
  const float* src = outs + ((size_t)n * 2 + c) * 4096;
  const float v00 = src[y0 * 64 + x0], v01 = src[y0 * 64 + x1];
  const float v10 = src[y1 * 64 + x0], v11 = src[y1 * 64 + x1];
  const float top = v00 + (v01 - v00) * wx;
  const float bot = v10 + (v11 - v10) * wx;
  oup[idx] = top + (bot - top) * wy;
}

// ---------------------------------------------------------------------------
// Kernel 3: per-pixel 81-tap correlation + online softmax + aggregation.
// 16x16 pixel tile per block; 24x24 reflect halo of f2 (16ch) and out_up (2ch)
// staged in LDS (41.5 KB of the 320 KB WGP pool).  Single pass, O(1) regs/px.
// Row stride 24 vs 64 LDS banks -> the two rows a wave touches are disjoint
// bank ranges (no conflicts).
// ---------------------------------------------------------------------------
__global__ __launch_bounds__(256) void k_attn(
    const float* __restrict__ f1, const float* __restrict__ f2,
    const float* __restrict__ oup, const float* __restrict__ ab,
    const float* __restrict__ gd_p, const float* __restrict__ gs_p,
    float* __restrict__ outp) {
  __shared__ float sf2[CMID * 576];   // [c][24*24]
  __shared__ float so[2 * 576];       // [c][24*24]

  const int tid = threadIdx.x;
  const int bx = blockIdx.x & 7;
  const int by = (blockIdx.x >> 3) & 7;
  const int n  = blockIdx.x >> 6;
  const int x0 = bx * 16, y0 = by * 16;

  // Stage reflect-padded halo.
  const float* f2n = f2  + (size_t)n * CMID * HWp;
  const float* on  = oup + (size_t)n * 2 * HWp;
  for (int i = tid; i < 576; i += 256) {
    const int ly = i / 24, lx = i % 24;
    int gy = y0 - 4 + ly; gy = gy < 0 ? -gy : (gy > 127 ? 254 - gy : gy);
    int gx = x0 - 4 + lx; gx = gx < 0 ? -gx : (gx > 127 ? 254 - gx : gx);
    const int g = gy * Ww + gx;
#pragma unroll
    for (int c = 0; c < CMID; ++c) sf2[c * 576 + i] = f2n[(size_t)c * HWp + g];
    so[i]       = on[g];
    so[576 + i] = on[HWp + g];
  }
  __syncthreads();

  const int tx = tid & 15, ty = tid >> 4;
  const int x = x0 + tx, y = y0 + ty;
  const int p = y * Ww + x;

  float f1c[CMID];
  const float* f1n = f1 + (size_t)n * CMID * HWp + p;
#pragma unroll
  for (int c = 0; c < CMID; ++c) f1c[c] = f1n[(size_t)c * HWp];

  const float gd = gd_p[0];
  const float gs = gs_p[0];
  const float* abn = ab + (size_t)n * 81 * HWp + p;   // coalesced over tx

  float mrun = -3.0e38f, ssum = 0.0f, a0 = 0.0f, a1 = 0.0f;
  for (int k = 0; k < 81; ++k) {
    const int dy = k / 9, dx = k % 9;                 // 0..8 (halo-shifted)
    const int off = (ty + dy) * 24 + (tx + dx);
    float dist = 0.0f;
#pragma unroll
    for (int c = 0; c < CMID; ++c) {
      const float d = f1c[c] - sf2[c * 576 + off];
      dist = fmaf(d, d, dist);
    }
    const float l  = fmaf(gs, abn[(size_t)k * HWp], -gd * dist);
    const float mn = fmaxf(mrun, l);
    const float sc = __expf(mrun - mn);               // first iter: exp(-inf)=0
    const float e  = __expf(l - mn);
    ssum = fmaf(ssum, sc, e);
    a0   = fmaf(a0, sc, e * so[off]);
    a1   = fmaf(a1, sc, e * so[576 + off]);
    mrun = mn;
  }
  const float inv = 1.0f / ssum;
  float* op = outp + (size_t)n * 2 * HWp + p;
  op[0]   = a0 * inv;
  op[HWp] = a1 * inv;
}

// ---------------------------------------------------------------------------
extern "C" void kernel_launch(void* const* d_in, const int* in_sizes, int n_in,
                              void* d_out, int out_size, void* d_ws, size_t ws_size,
                              hipStream_t stream) {
  const float* delta = (const float*)d_in[0];
  const float* outs  = (const float*)d_in[1];
  const float* ab    = (const float*)d_in[2];
  const float* w1    = (const float*)d_in[3];
  const float* b1    = (const float*)d_in[4];
  const float* w2    = (const float*)d_in[5];
  const float* b2    = (const float*)d_in[6];
  const float* gd    = (const float*)d_in[7];
  const float* gs    = (const float*)d_in[8];

  // Workspace layout (floats): f1 | f2 | out_up   (~8.6 MB total, L2-resident)
  float* f1  = (float*)d_ws;
  float* f2  = f1 + (size_t)N_ * CMID * HWp;   // 1,048,576 floats
  float* oup = f2 + (size_t)N_ * CMID * HWp;   // +131,072 floats

  // 1024 wave-jobs (64 px each) / 8 waves per block
  k_gemm<<<128, 256, 0, stream>>>(delta, w1, b1, w2, b2, f1, f2);
  // 131072 elements
  k_upsample<<<512, 256, 0, stream>>>(outs, oup);
  // 8x8 tiles x 4 batches
  k_attn<<<256, 256, 0, stream>>>(f1, f2, oup, ab, gd, gs, (float*)d_out);
}
